// MQA_953482739764
// MI455X (gfx1250) — compile-verified, matrix operand order checked by reading, not run on hardware
//
#include <hip/hip_runtime.h>

// ---------------------------------------------------------------------------
// MQA forward for MI455X (gfx1250, wave32, WMMA, async global->LDS).
// Pipeline: [q/k/v proj bf16-WMMA] -> [flash attention bf16-WMMA] -> [o proj]
// ---------------------------------------------------------------------------

typedef __bf16 bf16;
typedef __bf16 v16bf __attribute__((ext_vector_type(16)));
typedef __bf16 v8bf  __attribute__((ext_vector_type(8)));
typedef __bf16 v4bf  __attribute__((ext_vector_type(4)));
typedef float  v8f   __attribute__((ext_vector_type(8)));

#define NB    4
#define NS    2048
#define NHID  2048
#define NHEAD 16
#define HD    128

static __device__ inline v8f wmma_bf16(v16bf a, v16bf b, v8f c) {
  // (neg_a, A, neg_b, B, c_mod, C, reuse_a, reuse_b)
  return __builtin_amdgcn_wmma_f32_16x16x32_bf16(false, a, false, b,
                                                 (short)0, c, false, false);
}

static __device__ inline v16bf cat8(v8bf lo, v8bf hi) {
  return __builtin_shufflevector(lo, hi, 0, 1, 2, 3, 4, 5, 6, 7,
                                         8, 9, 10, 11, 12, 13, 14, 15);
}

// CDNA5 async copy: 16B global -> LDS per lane, tracked by ASYNCcnt.
static __device__ inline void async_copy_b128(const bf16* g, bf16* l) {
  unsigned loff = (unsigned)(unsigned long long)l;  // flat low 32b = LDS offset
  unsigned long long ga = (unsigned long long)g;
  asm volatile("global_load_async_to_lds_b128 %0, %1, off"
               :: "v"(loff), "v"(ga) : "memory");
}
static __device__ inline void wait_async0() {
  asm volatile("s_wait_asynccnt 0" ::: "memory");
}

// Stage 8 elements into LDS as bf16 (converting f32 on the fly).
static __device__ inline void stage8(const float* __restrict__ p, bf16* dst) {
  float4 a = *(const float4*)p;
  float4 b = *(const float4*)(p + 4);
  v8bf o = {(bf16)a.x, (bf16)a.y, (bf16)a.z, (bf16)a.w,
            (bf16)b.x, (bf16)b.y, (bf16)b.z, (bf16)b.w};
  *(v8bf*)dst = o;
}
static __device__ inline void stage8(const bf16* __restrict__ p, bf16* dst) {
  *(v8bf*)dst = *(const v8bf*)p;
}

// ---------------------------------------------------------------------------
// C[M,N] = A[M,K] @ W[K,N] + bias ; A is f32 or bf16, W/bias f32.
// Block tile 128x128, 256 threads = 8 waves (4M x 2N); each wave owns 32x64 =
// 2x4 WMMA accumulators. K-step 32. Double-buffered LDS (one barrier/step).
// B tile stored transposed [n][k]; 4 consecutive k packed -> b64 LDS stores.
// ---------------------------------------------------------------------------
template <typename AT, bool STORE_F32>
__global__ __launch_bounds__(256) void gemm_bias(
    const AT* __restrict__ A, const float* __restrict__ W,
    const float* __restrict__ bias, void* __restrict__ Out,
    int M, int N, int K) {
  __shared__ alignas(16) bf16 lds_a[2][128][32];  // [m][k]  16 KB
  __shared__ alignas(16) bf16 lds_b[2][128][32];  // [n][k]  16 KB

  const int tid  = threadIdx.x;
  const int lane = tid & 31;
  const int wave = tid >> 5;
  const int half = lane >> 4;
  const int l16  = lane & 15;
  const int m0   = blockIdx.y * 128;
  const int n0   = blockIdx.x * 128;
  const int wm   = (wave >> 1) * 32;
  const int wn   = (wave & 1) * 64;

  auto load_tiles = [&](int kb, int buf) {
    // A tile: 128x32, coalesced 16B/32B segments, converted in flight.
#pragma unroll
    for (int it = 0; it < 2; ++it) {
      int lin = tid + it * 256;
      int r = lin >> 2, kc = (lin & 3) * 8;
      stage8(A + (m0 + r) * K + kb + kc, &lds_a[buf][r][kc]);
    }
    // B tile: 4 k-rows of W read coalesced, packed into b64 transposed stores.
    int kq = tid >> 5, ncc = (tid & 31) * 4;
    float wf[4][4];
#pragma unroll
    for (int k2 = 0; k2 < 4; ++k2) {
      const float4 f = *(const float4*)&W[(kb + kq * 4 + k2) * N + n0 + ncc];
      wf[k2][0] = f.x; wf[k2][1] = f.y; wf[k2][2] = f.z; wf[k2][3] = f.w;
    }
#pragma unroll
    for (int j = 0; j < 4; ++j) {
      v4bf p = {(bf16)wf[0][j], (bf16)wf[1][j], (bf16)wf[2][j], (bf16)wf[3][j]};
      *(v4bf*)&lds_b[buf][ncc + j][kq * 4] = p;
    }
  };

  const v8f z8 = {0.f, 0.f, 0.f, 0.f, 0.f, 0.f, 0.f, 0.f};
  v8f acc[2][4];
#pragma unroll
  for (int mt = 0; mt < 2; ++mt)
#pragma unroll
    for (int nt = 0; nt < 4; ++nt) acc[mt][nt] = z8;

  load_tiles(0, 0);
  const int KSTEPS = K / 32;
  for (int i = 0; i < KSTEPS; ++i) {
    __syncthreads();  // buf(i) ready; buf(i^1) free to overwrite
    if (i + 1 < KSTEPS) load_tiles((i + 1) * 32, (i + 1) & 1);
    const int buf = i & 1;
#pragma unroll
    for (int mt = 0; mt < 2; ++mt) {
      int ar = wm + mt * 16 + l16;
      v16bf a = cat8(*(const v8bf*)&lds_a[buf][ar][half * 8],
                     *(const v8bf*)&lds_a[buf][ar][16 + half * 8]);
#pragma unroll
      for (int nt = 0; nt < 4; ++nt) {
        int bc = wn + nt * 16 + l16;
        v16bf b = cat8(*(const v8bf*)&lds_b[buf][bc][half * 8],
                       *(const v8bf*)&lds_b[buf][bc][16 + half * 8]);
        acc[mt][nt] = wmma_bf16(a, b, acc[mt][nt]);
      }
    }
  }

  // Epilogue: C/D layout VGPR r -> row (r + 8*half), col = lane&15.
#pragma unroll
  for (int mt = 0; mt < 2; ++mt)
#pragma unroll
    for (int nt = 0; nt < 4; ++nt) {
      int col = n0 + wn + nt * 16 + l16;
      float bv = bias[col];
#pragma unroll
      for (int r = 0; r < 8; ++r) {
        int row = m0 + wm + mt * 16 + half * 8 + r;
        float v = acc[mt][nt][r] + bv;
        if (STORE_F32)
          ((float*)Out)[row * N + col] = v;
        else
          ((bf16*)Out)[row * N + col] = (bf16)v;
      }
    }
}

// ---------------------------------------------------------------------------
// Flash attention, MQA (1 KV head). Block = 64 q rows of one (b,h); 4 waves,
// 16 q-rows each. Online softmax; scores stay in-WGP. Q/K tiles fetched with
// GLOBAL_LOAD_ASYNC_TO_LDS_B128 (ASYNCcnt); K/V double-buffered. Output is
// written pre-permuted as [B, NH*HD, S] bf16 == row-major [B*S, H] for the
// final projection (this *is* the reference's swapaxes(-1,-2)+reshape).
// ---------------------------------------------------------------------------
__global__ __launch_bounds__(128) void mqa_attn(
    const bf16* __restrict__ Qm, const bf16* __restrict__ Km,
    const bf16* __restrict__ Vm, bf16* __restrict__ Op) {
  __shared__ alignas(16) bf16 lds_q[64][128];      // [s][hd]       16 KB
  __shared__ alignas(16) bf16 lds_k[2][64][128];   // [key][hd]     32 KB
  __shared__ alignas(16) bf16 lds_vt[2][128][64];  // [hd][key]     32 KB
  __shared__ alignas(16) bf16 lds_p[4][16][64];    // per-wave probs 8 KB

  const int tid  = threadIdx.x;
  const int lane = tid & 31;
  const int wave = tid >> 5;
  const int half = lane >> 4;
  const int l16  = lane & 15;
  const int b    = blockIdx.x / NHEAD;
  const int h    = blockIdx.x % NHEAD;
  const int q0   = blockIdx.y * 64;
  const float scale = 0.08838834764831845f;  // 1/sqrt(128)

  // ---- prologue: async Q tile + async K tile(0); V tile(0) via VGPR ----
#pragma unroll
  for (int it = 0; it < 8; ++it) {
    int lin = tid + it * 128;
    int r = lin >> 4, cc = (lin & 15) * 8;
    async_copy_b128(&Qm[(b * NS + q0 + r) * NHID + h * HD + cc],
                    &lds_q[r][cc]);
    async_copy_b128(&Km[(b * NS + r) * HD + cc], &lds_k[0][r][cc]);
  }
#pragma unroll
  for (int g = 0; g < 2; ++g) {
    int lin = tid + g * 128;
    int rq = lin >> 4, cc = (lin & 15) * 8;
    v8bf vv[4];
#pragma unroll
    for (int k2 = 0; k2 < 4; ++k2)
      vv[k2] = *(const v8bf*)&Vm[(b * NS + rq * 4 + k2) * HD + cc];
#pragma unroll
    for (int j = 0; j < 8; ++j) {
      v4bf p = {vv[0][j], vv[1][j], vv[2][j], vv[3][j]};
      *(v4bf*)&lds_vt[0][cc + j][rq * 4] = p;  // 4 keys packed per b64 store
    }
  }
  wait_async0();
  __syncthreads();

  const v8f z8 = {0.f, 0.f, 0.f, 0.f, 0.f, 0.f, 0.f, 0.f};
  v8f oacc[8];
#pragma unroll
  for (int t = 0; t < 8; ++t) oacc[t] = z8;
  float m_run[8], l_run[8];
#pragma unroll
  for (int r = 0; r < 8; ++r) { m_run[r] = -1e30f; l_run[r] = 0.f; }

  const int NBLK = NS / 64;
  for (int i = 0; i < NBLK; ++i) {
    const int cur = i & 1, nxt = cur ^ 1;
    const int kb = i * 64;

    // Issue next tile's loads before computing on the current one.
    v8bf vreg[8];
    const bool more = (i + 1) < NBLK;
    if (more) {
      const int kn = kb + 64;
#pragma unroll
      for (int it = 0; it < 8; ++it) {
        int lin = tid + it * 128;
        int r = lin >> 4, cc = (lin & 15) * 8;
        async_copy_b128(&Km[(b * NS + kn + r) * HD + cc], &lds_k[nxt][r][cc]);
      }
#pragma unroll
      for (int g = 0; g < 2; ++g) {
        int lin = tid + g * 128;
        int rq = lin >> 4, cc = (lin & 15) * 8;
#pragma unroll
        for (int k2 = 0; k2 < 4; ++k2)
          vreg[g * 4 + k2] =
              *(const v8bf*)&Vm[(b * NS + kn + rq * 4 + k2) * HD + cc];
      }
    }

    // S_blk [16 q x 64 key] = Q_tile @ K_tile^T (K-dim = HD = 4 wmma steps)
    v8f sacc[4];
#pragma unroll
    for (int nt = 0; nt < 4; ++nt) sacc[nt] = z8;
#pragma unroll
    for (int kc = 0; kc < 4; ++kc) {
      int qr = wave * 16 + l16;
      v16bf a = cat8(*(const v8bf*)&lds_q[qr][kc * 32 + half * 8],
                     *(const v8bf*)&lds_q[qr][kc * 32 + 16 + half * 8]);
#pragma unroll
      for (int nt = 0; nt < 4; ++nt) {
        int key = nt * 16 + l16;
        v16bf bb = cat8(*(const v8bf*)&lds_k[cur][key][kc * 32 + half * 8],
                        *(const v8bf*)&lds_k[cur][key][kc * 32 + 16 + half * 8]);
        sacc[nt] = wmma_bf16(a, bb, sacc[nt]);
      }
    }
#pragma unroll
    for (int nt = 0; nt < 4; ++nt)
#pragma unroll
      for (int r = 0; r < 8; ++r) sacc[nt][r] *= scale;

    // Online softmax. D layout: element r belongs to row r+8*half; the row's
    // 16 columns live across this half's lanes -> xor-butterfly 8..1.
#pragma unroll
    for (int r = 0; r < 8; ++r) {
      float mx = sacc[0][r];
#pragma unroll
      for (int nt = 1; nt < 4; ++nt) mx = fmaxf(mx, sacc[nt][r]);
#pragma unroll
      for (int off = 8; off >= 1; off >>= 1)
        mx = fmaxf(mx, __shfl_xor(mx, off, 32));
      float mnew = fmaxf(m_run[r], mx);
      float alpha = __expf(m_run[r] - mnew);
      m_run[r] = mnew;
      float s = 0.f;
#pragma unroll
      for (int nt = 0; nt < 4; ++nt) {
        float p = __expf(sacc[nt][r] - mnew);
        sacc[nt][r] = p;
        s += p;
      }
#pragma unroll
      for (int off = 8; off >= 1; off >>= 1) s += __shfl_xor(s, off, 32);
      l_run[r] = l_run[r] * alpha + s;
#pragma unroll
      for (int t = 0; t < 8; ++t) oacc[t][r] *= alpha;
    }

    // D-layout -> A-layout via per-wave LDS round trip.
#pragma unroll
    for (int nt = 0; nt < 4; ++nt)
#pragma unroll
      for (int r = 0; r < 8; ++r)
        lds_p[wave][half * 8 + r][nt * 16 + l16] = (bf16)sacc[nt][r];

    // O += P @ V_blk (K-dim = 64 keys = 2 wmma steps, 8 hd n-tiles)
#pragma unroll
    for (int kc = 0; kc < 2; ++kc) {
      v16bf a = cat8(*(const v8bf*)&lds_p[wave][l16][kc * 32 + half * 8],
                     *(const v8bf*)&lds_p[wave][l16][kc * 32 + 16 + half * 8]);
#pragma unroll
      for (int t = 0; t < 8; ++t) {
        int hd = t * 16 + l16;
        v16bf bb = cat8(*(const v8bf*)&lds_vt[cur][hd][kc * 32 + half * 8],
                        *(const v8bf*)&lds_vt[cur][hd][kc * 32 + 16 + half * 8]);
        oacc[t] = wmma_bf16(a, bb, oacc[t]);
      }
    }

    // Drain next V tile into its transposed LDS buffer.
    if (more) {
#pragma unroll
      for (int g = 0; g < 2; ++g) {
        int lin = tid + g * 128;
        int rq = lin >> 4, cc = (lin & 15) * 8;
#pragma unroll
        for (int j = 0; j < 8; ++j) {
          v4bf p = {vreg[g * 4 + 0][j], vreg[g * 4 + 1][j],
                    vreg[g * 4 + 2][j], vreg[g * 4 + 3][j]};
          *(v4bf*)&lds_vt[nxt][cc + j][rq * 4] = p;
        }
      }
    }
    wait_async0();     // own async K writes landed
    __syncthreads();   // everyone's next-tile writes visible; cur now free
  }

  __syncthreads();  // done reading lds_q; reuse it for the output transpose
#pragma unroll
  for (int r = 0; r < 8; ++r) {
    float inv = 1.f / l_run[r];
    int sl = wave * 16 + half * 8 + r;
#pragma unroll
    for (int t = 0; t < 8; ++t)
      lds_q[sl][t * 16 + l16] = (bf16)(oacc[t][r] * inv);
  }
  __syncthreads();

  // Reference permutation: flat-in-batch index (h*HD + hd)*S + s.
  // Consecutive tid 0..63 write consecutive s -> coalesced 128B runs.
  const int s_l  = tid & 63;
  const int hd_o = tid >> 6;
  for (int hb = 0; hb < 64; ++hb) {
    int hd = hb * 2 + hd_o;
    Op[((b * NHEAD + h) * HD + hd) * NS + q0 + s_l] = lds_q[s_l][hd];
  }
}

// ---------------------------------------------------------------------------
extern "C" void kernel_launch(void* const* d_in, const int* in_sizes, int n_in,
                              void* d_out, int out_size, void* d_ws,
                              size_t ws_size, hipStream_t stream) {
  (void)in_sizes; (void)n_in; (void)out_size; (void)ws_size;
  const float* hs  = (const float*)d_in[0];
  const float* w_q = (const float*)d_in[1];
  const float* b_q = (const float*)d_in[2];
  const float* w_k = (const float*)d_in[3];
  const float* b_k = (const float*)d_in[4];
  const float* w_v = (const float*)d_in[5];
  const float* b_v = (const float*)d_in[6];
  const float* w_o = (const float*)d_in[7];
  const float* b_o = (const float*)d_in[8];

  const int M = NB * NS;  // 8192
  char* ws = (char*)d_ws;
  bf16* q_bf = (bf16*)ws;                                     // 32 MB
  bf16* k_bf = (bf16*)(ws + (size_t)M * NHID * 2);            //  2 MB
  bf16* v_bf = (bf16*)(ws + (size_t)M * NHID * 2 + (size_t)M * HD * 2);
  bf16* attn = (bf16*)(ws + (size_t)M * NHID * 2 + 2 * (size_t)M * HD * 2);

  dim3 blk(256);
  gemm_bias<float, false><<<dim3(NHID / 128, M / 128), blk, 0, stream>>>(
      hs, w_q, b_q, q_bf, M, NHID, NHID);
  gemm_bias<float, false><<<dim3(HD / 128, M / 128), blk, 0, stream>>>(
      hs, w_k, b_k, k_bf, M, HD, NHID);
  gemm_bias<float, false><<<dim3(HD / 128, M / 128), blk, 0, stream>>>(
      hs, w_v, b_v, v_bf, M, HD, NHID);
  mqa_attn<<<dim3(NB * NHEAD, NS / 64), dim3(128), 0, stream>>>(
      q_bf, k_bf, v_bf, attn);
  gemm_bias<bf16, true><<<dim3(NHID / 128, M / 128), blk, 0, stream>>>(
      attn, w_o, b_o, d_out, M, NHID, NHID);
}
